// AILSTM_45097156608139
// MI455X (gfx1250) — matrix operand order repaired and checked by reading, compiled, not run on hardware
//
#include <hip/hip_runtime.h>
#include <hip/hip_bf16.h>
#include <math.h>

// ---------------------------------------------------------------------------
// Problem constants (from reference): B=4, L=512, EMBED=300, H=128, 4H=512.
// Row space for "stacked" activations: row = s*2048 + b*512 + t, s in {0=q,1=a}.
// ---------------------------------------------------------------------------
#define NROWS 4096      // 2 * B * L
#define KPAD  320       // EMBED padded to multiple of 32
#define G4H   512       // 4*H
#define HDIM  128

typedef __attribute__((ext_vector_type(16))) __bf16 bf16x16;
typedef __attribute__((ext_vector_type(8)))  __bf16 bf16x8;
typedef __attribute__((ext_vector_type(8)))  float  f32x8;

static __device__ __forceinline__ bf16x16 pack16(bf16x8 lo, bf16x8 hi) {
  return __builtin_shufflevector(lo, hi, 0,1,2,3,4,5,6,7,8,9,10,11,12,13,14,15);
}

// ---------------------------------------------------------------------------
// Embedding gather -> bf16, K padded 300->320 with zeros.
// X is (4096 x 320) bf16 row-major; rows 0..2047 = question, 2048..4095 = answer.
// ---------------------------------------------------------------------------
__global__ void k_embed_bf16(const int* __restrict__ qc, const int* __restrict__ ac,
                             const float* __restrict__ emb, __bf16* __restrict__ X) {
  size_t idx = (size_t)blockIdx.x * blockDim.x + threadIdx.x;
  if (idx >= (size_t)NROWS * KPAD) return;
  int k  = (int)(idx % KPAD);
  int r  = (int)(idx / KPAD);
  int s  = r >> 11;           // 0 = question, 1 = answer
  int bl = r & 2047;          // b*512 + t  (matches flat (B,L) index)
  int tok = (s == 0 ? qc : ac)[bl];
  float v = (k < 300) ? emb[(size_t)tok * 300 + k] : 0.0f;
  X[idx] = (__bf16)v;
}

// ---------------------------------------------------------------------------
// Transpose + convert f32 (K x N) -> bf16 (N x Kp), zero-padding k >= K.
// Row length of input is N.
// ---------------------------------------------------------------------------
__global__ void k_transpose_bf16(const float* __restrict__ in, __bf16* __restrict__ out,
                                 int K, int N, int Kp) {
  size_t idx = (size_t)blockIdx.x * blockDim.x + threadIdx.x;
  if (idx >= (size_t)N * Kp) return;
  int k = (int)(idx % Kp);
  int n = (int)(idx / Kp);
  float v = (k < K) ? in[(size_t)k * N + n] : 0.0f;
  out[idx] = (__bf16)v;
}

// ---------------------------------------------------------------------------
// WMMA GEMM: C[M,N] (f32) = A[M,K](bf16, row-major) * B, with B given as
// BT[N,K] (bf16, row-major), optional bias[N].
// One wave computes a 16x32 output tile (two accumulators share one A frag,
// giving two independent WMMA chains per k-step for latency hiding).
// K must be a multiple of 32; M multiple of 16; N multiple of 32.
// ---------------------------------------------------------------------------
__global__ __launch_bounds__(32) void k_gemm_bf16(const __bf16* __restrict__ A,
                                                  const __bf16* __restrict__ BT,
                                                  const float* __restrict__ bias,
                                                  float* __restrict__ C,
                                                  int M, int N, int K) {
  const int tm   = blockIdx.x * 16;
  const int tn   = blockIdx.y * 32;
  const int lane = threadIdx.x & 31;
  const int half = lane >> 4;
  const int l15  = lane & 15;

  const __bf16* ap  = A  + (size_t)(tm + l15) * K + half * 16;       // A: contiguous 16 K at half*16
  const __bf16* bp0 = BT + (size_t)(tn + l15) * K + half * 16;       // B: contiguous 16 K at half*16
  const __bf16* bp1 = BT + (size_t)(tn + 16 + l15) * K + half * 16;

  f32x8 acc0 = {}, acc1 = {};
  for (int k0 = 0; k0 < K; k0 += 32) {
    bf16x8 alo  = *(const bf16x8*)(ap + k0);
    bf16x8 ahi  = *(const bf16x8*)(ap + k0 + 8);
    bf16x8 b0lo = *(const bf16x8*)(bp0 + k0);
    bf16x8 b0hi = *(const bf16x8*)(bp0 + k0 + 8);
    bf16x8 b1lo = *(const bf16x8*)(bp1 + k0);
    bf16x8 b1hi = *(const bf16x8*)(bp1 + k0 + 8);
    bf16x16 a  = pack16(alo, ahi);
    bf16x16 b0 = pack16(b0lo, b0hi);
    bf16x16 b1 = pack16(b1lo, b1hi);
    acc0 = __builtin_amdgcn_wmma_f32_16x16x32_bf16(false, a, false, b0,
                                                   (short)0, acc0, false, false);
    acc1 = __builtin_amdgcn_wmma_f32_16x16x32_bf16(false, a, false, b1,
                                                   (short)0, acc1, false, false);
  }
  const float bv0 = bias ? bias[tn + l15] : 0.0f;
  const float bv1 = bias ? bias[tn + 16 + l15] : 0.0f;
#pragma unroll
  for (int r = 0; r < 8; ++r) {
    int row = tm + r + 8 * half;
    C[(size_t)row * N + tn + l15]      = acc0[r] + bv0;
    C[(size_t)row * N + tn + 16 + l15] = acc1[r] + bv1;
  }
}

// ---------------------------------------------------------------------------
// LSTM recurrence. One persistent block (32 waves) per LSTM (blockIdx.x: 0=q,1=a).
// Gx = precomputed x@Wih + b  (4096 x 512 f32). WhhT = (512 x 128) bf16.
// h lives in LDS as a 16x128 bf16 matrix (rows 0..3 = batch, 4..15 = zero).
// The loop-invariant Whh B-fragments are preloaded into registers (32 VGPRs),
// so each serial step is just: LDS h-frag loads -> 4 x WMMA -> barrier.
// ---------------------------------------------------------------------------
__global__ __launch_bounds__(1024) void k_lstm(const float* __restrict__ Gx,
                                               const __bf16* __restrict__ WhhT,
                                               float* __restrict__ hsf,
                                               __bf16* __restrict__ hsbf) {
  const int s    = blockIdx.x;
  const int tid  = threadIdx.x;
  const int wave = tid >> 5;
  const int lane = tid & 31;
  const int half = lane >> 4;
  const int l15  = lane & 15;
  const int n0   = wave * 16;

  __shared__ __align__(16) __bf16 hbf[16 * HDIM];   // padded h (bf16)
  __shared__ float cst[4 * HDIM];                   // cell state
  __shared__ float gates[4 * G4H];                  // pre-activation gates

  for (int i = tid; i < 16 * HDIM; i += 1024) hbf[i] = (__bf16)0.0f;
  for (int i = tid; i < 4 * HDIM;  i += 1024) cst[i] = 0.0f;

  // Preload the recurrent weights for this wave's 16-column gate tile.
  const __bf16* bbase = WhhT + (size_t)(n0 + l15) * HDIM + half * 16;
  bf16x16 bfrag[4];
#pragma unroll
  for (int kt = 0; kt < 4; ++kt) {
    bf16x8 blo = *(const bf16x8*)(bbase + kt * 32);
    bf16x8 bhi = *(const bf16x8*)(bbase + kt * 32 + 8);
    bfrag[kt] = pack16(blo, bhi);
  }
  __syncthreads();

  const size_t rowbase = (size_t)s * 2048;

  for (int t = 0; t < 512; ++t) {
    f32x8 acc = {};
#pragma unroll
    for (int kt = 0; kt < 4; ++kt) {
      const __bf16* hp = &hbf[l15 * HDIM + kt * 32 + half * 16];
      bf16x8 alo = *(const bf16x8*)hp;
      bf16x8 ahi = *(const bf16x8*)(hp + 8);
      bf16x16 a = pack16(alo, ahi);
      acc = __builtin_amdgcn_wmma_f32_16x16x32_bf16(false, a, false, bfrag[kt],
                                                    (short)0, acc, false, false);
    }
    // Only batch rows 0..3 are live: lanes 0..15 (half==0), accumulators 0..3.
    if (half == 0) {
#pragma unroll
      for (int r = 0; r < 4; ++r) {
        size_t grow = rowbase + (size_t)r * 512 + t;
        gates[r * G4H + n0 + l15] = acc[r] + Gx[grow * G4H + n0 + l15];
      }
    }
    __syncthreads();
    if (tid < 512) {
      const int b  = tid >> 7;
      const int hh = tid & (HDIM - 1);
      float gi = gates[b * G4H + hh];
      float gf = gates[b * G4H + HDIM + hh];
      float gg = gates[b * G4H + 2 * HDIM + hh];
      float go = gates[b * G4H + 3 * HDIM + hh];
      float si = 1.0f / (1.0f + expf(-gi));
      float sf = 1.0f / (1.0f + expf(-gf));
      float so = 1.0f / (1.0f + expf(-go));
      float cv = sf * cst[b * HDIM + hh] + si * tanhf(gg);
      cst[b * HDIM + hh] = cv;
      float hv = so * tanhf(cv);
      hbf[b * HDIM + hh] = (__bf16)hv;
      size_t row = rowbase + (size_t)b * 512 + t;
      hsf[row * HDIM + hh]  = hv;
      hsbf[row * HDIM + hh] = (__bf16)hv;
    }
    __syncthreads();
  }
}

// ---------------------------------------------------------------------------
// Attention tail. tanh is monotonic, so
//   r_q[b,j,h] = tanh(max_i proj_q[b,i,h] + proj_a[b,j,h] + ba[h])
//   r_a[b,i,h] = tanh(proj_q[b,i,h] + max_j proj_a[b,j,h] + ba[h])
// which collapses the 512x512x128 interaction tensor to O(L*H).
// One block per batch element.
// ---------------------------------------------------------------------------
__global__ __launch_bounds__(256) void k_final(const float* __restrict__ proj,
                                               const float* __restrict__ hsf,
                                               const float* __restrict__ ba,
                                               const float* __restrict__ Wq,
                                               const float* __restrict__ bq,
                                               const float* __restrict__ Wans,
                                               const float* __restrict__ bans,
                                               const float* __restrict__ Wlast,
                                               const float* __restrict__ blast,
                                               float* __restrict__ out) {
  const int b   = blockIdx.x;
  const int tid = threadIdx.x;
  const float* pq = proj + (size_t)(b * 512) * HDIM;
  const float* pa = proj + (size_t)(2048 + b * 512) * HDIM;
  const float* hq = hsf  + (size_t)(b * 512) * HDIM;
  const float* ha = hsf  + (size_t)(2048 + b * 512) * HDIM;

  __shared__ float mq[HDIM], ma[HDIM];
  __shared__ float sq[512], sa[512];
  __shared__ float red[256];
  __shared__ float featq[HDIM], feata[HDIM];

  // --- max over positions ---
  if (tid < HDIM) {
    float m = -3.4e38f;
    for (int p = 0; p < 512; ++p) m = fmaxf(m, pq[(size_t)p * HDIM + tid]);
    mq[tid] = m;
  } else {
    const int h = tid - HDIM;
    float m = -3.4e38f;
    for (int p = 0; p < 512; ++p) m = fmaxf(m, pa[(size_t)p * HDIM + h]);
    ma[h] = m;
  }
  __syncthreads();

  // --- attention scores s_q[p], s_a[p] ---
  for (int p = tid; p < 512; p += 256) {
    float s1 = 0.0f, s2 = 0.0f;
    for (int h = 0; h < HDIM; ++h) {
      float paV = pa[(size_t)p * HDIM + h];
      float pqV = pq[(size_t)p * HDIM + h];
      float rq = tanhf(mq[h] + paV + ba[h]);
      float ra = tanhf(pqV + ma[h] + ba[h]);
      s1 += hq[(size_t)p * HDIM + h] * Wq[h]   + rq * Wq[HDIM + h];
      s2 += ha[(size_t)p * HDIM + h] * Wans[h] + ra * Wans[HDIM + h];
    }
    sq[p] = s1 + bq[0];
    sa[p] = s2 + bans[0];
  }
  __syncthreads();

  // --- softmax over 512 positions, in place, for sq then sa ---
  for (int pass = 0; pass < 2; ++pass) {
    float* sv = pass == 0 ? sq : sa;
    red[tid] = fmaxf(sv[tid], sv[tid + 256]);
    __syncthreads();
    for (int st = 128; st > 0; st >>= 1) {
      if (tid < st) red[tid] = fmaxf(red[tid], red[tid + st]);
      __syncthreads();
    }
    float mx = red[0];
    __syncthreads();
    float e0 = expf(sv[tid] - mx), e1 = expf(sv[tid + 256] - mx);
    sv[tid] = e0; sv[tid + 256] = e1;
    red[tid] = e0 + e1;
    __syncthreads();
    for (int st = 128; st > 0; st >>= 1) {
      if (tid < st) red[tid] += red[tid + st];
      __syncthreads();
    }
    float inv = 1.0f / red[0];
    __syncthreads();
    sv[tid] *= inv; sv[tid + 256] *= inv;
    __syncthreads();
  }

  // --- pooled features ---
  if (tid < HDIM) {
    float acc = 0.0f;
    for (int p = 0; p < 512; ++p)
      acc += sq[p] * tanhf(mq[tid] + pa[(size_t)p * HDIM + tid] + ba[tid]);
    featq[tid] = acc;
  } else {
    const int h = tid - HDIM;
    float acc = 0.0f;
    for (int p = 0; p < 512; ++p)
      acc += sa[p] * tanhf(pq[(size_t)p * HDIM + h] + ma[h] + ba[h]);
    feata[h] = acc;
  }
  __syncthreads();

  // --- classifier + log_softmax + argmax ---
  if (tid == 0) {
    float l0 = blast[0], l1 = blast[1];
    for (int h = 0; h < HDIM; ++h) {
      l0 += featq[h] * Wlast[h * 2 + 0] + feata[h] * Wlast[(HDIM + h) * 2 + 0];
      l1 += featq[h] * Wlast[h * 2 + 1] + feata[h] * Wlast[(HDIM + h) * 2 + 1];
    }
    float m  = fmaxf(l0, l1);
    float ls = m + logf(expf(l0 - m) + expf(l1 - m));
    out[b * 2 + 0] = l0 - ls;
    out[b * 2 + 1] = l1 - ls;
    out[8 + b] = (l1 > l0) ? 1.0f : 0.0f;   // argmax (first index wins ties)
  }
}

// ---------------------------------------------------------------------------
// Host-side orchestration.
// ---------------------------------------------------------------------------
extern "C" void kernel_launch(void* const* d_in, const int* in_sizes, int n_in,
                              void* d_out, int out_size, void* d_ws, size_t ws_size,
                              hipStream_t stream) {
  const int*   qc     = (const int*)  d_in[0];
  const int*   ac     = (const int*)  d_in[1];
  const float* emb    = (const float*)d_in[2];
  const float* Wih    = (const float*)d_in[3];
  const float* Whh    = (const float*)d_in[4];
  const float* b_lstm = (const float*)d_in[5];
  const float* Wa     = (const float*)d_in[6];
  const float* ba     = (const float*)d_in[7];
  const float* Wq     = (const float*)d_in[8];
  const float* bq     = (const float*)d_in[9];
  const float* Wans   = (const float*)d_in[10];
  const float* bans   = (const float*)d_in[11];
  const float* Wlast  = (const float*)d_in[12];
  const float* blast  = (const float*)d_in[13];
  float* out = (float*)d_out;
  char*  ws  = (char*)d_ws;

  // workspace layout (all offsets multiples of 256B; alignment holds for b128 loads)
  constexpr size_t OFF_X    = 0;                                     // 4096x320 bf16
  constexpr size_t SZ_X     = (size_t)NROWS * KPAD * 2;
  constexpr size_t OFF_WIHT = OFF_X + SZ_X;                          // 512x320 bf16
  constexpr size_t SZ_WIHT  = (size_t)G4H * KPAD * 2;
  constexpr size_t OFF_WHHT = OFF_WIHT + SZ_WIHT;                    // 512x128 bf16
  constexpr size_t SZ_WHHT  = (size_t)G4H * HDIM * 2;
  constexpr size_t OFF_WAQT = OFF_WHHT + SZ_WHHT;                    // 128x128 bf16
  constexpr size_t SZ_WAT   = (size_t)HDIM * HDIM * 2;
  constexpr size_t OFF_WAAT = OFF_WAQT + SZ_WAT;
  constexpr size_t OFF_GX   = OFF_WAAT + SZ_WAT;                     // 4096x512 f32
  constexpr size_t SZ_GX    = (size_t)NROWS * G4H * 4;
  constexpr size_t OFF_HSF  = OFF_GX + SZ_GX;                        // 4096x128 f32
  constexpr size_t SZ_HSF   = (size_t)NROWS * HDIM * 4;
  constexpr size_t OFF_HSBF = OFF_HSF + SZ_HSF;                      // 4096x128 bf16
  constexpr size_t SZ_HSBF  = (size_t)NROWS * HDIM * 2;
  constexpr size_t OFF_PROJ = OFF_HSBF + SZ_HSBF;                    // 4096x128 f32

  __bf16* X    = (__bf16*)(ws + OFF_X);
  __bf16* WihT = (__bf16*)(ws + OFF_WIHT);
  __bf16* WhhT = (__bf16*)(ws + OFF_WHHT);
  __bf16* WaqT = (__bf16*)(ws + OFF_WAQT);
  __bf16* WaaT = (__bf16*)(ws + OFF_WAAT);
  float*  Gx   = (float*) (ws + OFF_GX);
  float*  hsf  = (float*) (ws + OFF_HSF);
  __bf16* hsbf = (__bf16*)(ws + OFF_HSBF);
  float*  proj = (float*) (ws + OFF_PROJ);

  // 1) gather + convert activations and weights
  {
    size_t n = (size_t)NROWS * KPAD;
    k_embed_bf16<<<dim3((unsigned)((n + 255) / 256)), 256, 0, stream>>>(qc, ac, emb, X);
  }
  k_transpose_bf16<<<(G4H * KPAD + 255) / 256, 256, 0, stream>>>(Wih, WihT, 300, G4H, KPAD);
  k_transpose_bf16<<<(G4H * HDIM + 255) / 256, 256, 0, stream>>>(Whh, WhhT, HDIM, G4H, HDIM);
  k_transpose_bf16<<<(HDIM * HDIM + 255) / 256, 256, 0, stream>>>(Wa, WaqT, HDIM, HDIM, HDIM);
  k_transpose_bf16<<<(HDIM * HDIM + 255) / 256, 256, 0, stream>>>(Wa + (size_t)HDIM * HDIM,
                                                                  WaaT, HDIM, HDIM, HDIM);

  // 2) input GEMM (hoisted out of the time loop): Gx = X @ Wih + b
  {
    dim3 grid(NROWS / 16, G4H / 32);
    k_gemm_bf16<<<grid, 32, 0, stream>>>(X, WihT, b_lstm, Gx, NROWS, G4H, KPAD);
  }

  // 3) both LSTM recurrences in parallel (persistent blocks, WMMA per step)
  k_lstm<<<2, 1024, 0, stream>>>(Gx, WhhT, hsf, hsbf);

  // 4) projections proj_q = hs_q @ Wa[:H], proj_a = hs_a @ Wa[H:]
  {
    dim3 grid(2048 / 16, HDIM / 32);
    k_gemm_bf16<<<grid, 32, 0, stream>>>(hsbf, WaqT, nullptr, proj, 2048, HDIM, HDIM);
    k_gemm_bf16<<<grid, 32, 0, stream>>>(hsbf + (size_t)2048 * HDIM, WaaT, nullptr,
                                         proj + (size_t)2048 * HDIM, 2048, HDIM, HDIM);
  }

  // 5) attention tail + classifier
  k_final<<<4, 256, 0, stream>>>(proj, hsf, ba, Wq, bq, Wans, bans, Wlast, blast, out);
}
